// KMeans_16518444221246
// MI455X (gfx1250) — matrix operand order repaired and checked by reading, not compiled, not run on hardware
//
#include <hip/hip_runtime.h>
#include <hip/hip_bf16.h>
#include <stdint.h>

#define B_PTS 131072
#define D_DIM 256
#define K_CLU 1024
#define L_LAB 526
#define EPSV  1e-8f

#define TILE_M 64      // point rows per block (4 waves x 16 rows)
#define NTHR   128     // 4 waves

typedef __attribute__((ext_vector_type(16))) __bf16 v16bf;
typedef __attribute__((ext_vector_type(8)))  float  v8f;
typedef __attribute__((ext_vector_type(4)))  int    v4i;
typedef __attribute__((ext_vector_type(8)))  int    v8i;

struct Q2 { v4i a, b; };   // 32 bytes, two 16B halves of a v16bf fragment

// ---- fp32 -> bf16 round-to-nearest-even ----
static __device__ __forceinline__ uint32_t bf16_rne(float f) {
    uint32_t u = __float_as_uint(f);
    return (u + 0x7FFFu + ((u >> 16) & 1u)) >> 16;
}

// ---------------------------------------------------------------------
__global__ void zero_kernel(float* __restrict__ out, int n) {
    int i = blockIdx.x * blockDim.x + threadIdx.x;
    if (i < n) out[i] = 0.0f;
}

// per-centroid: ||c_k||^2 and ||running_labels_k||
__global__ __launch_bounds__(256) void norms_kernel(
        const float* __restrict__ centroids,
        const float* __restrict__ running_labels,
        float* __restrict__ c2, float* __restrict__ rln) {
    __shared__ float red[256];
    const int k = blockIdx.x, t = threadIdx.x;
    float v = centroids[(size_t)k * D_DIM + t];
    red[t] = v * v;
    __syncthreads();
    for (int off = 128; off > 0; off >>= 1) {
        if (t < off) red[t] += red[t + off];
        __syncthreads();
    }
    if (t == 0) c2[k] = red[0];
    __syncthreads();
    float s = 0.f;
    for (int i = t; i < L_LAB; i += 256) {
        float w = running_labels[(size_t)k * L_LAB + i];
        s += w * w;
    }
    red[t] = s;
    __syncthreads();
    for (int off = 128; off > 0; off >>= 1) {
        if (t < off) red[t] += red[t + off];
        __syncthreads();
    }
    if (t == 0) rln[k] = sqrtf(red[0]);
}

// ---------------------------------------------------------------------
// WMMA distance + argmin kernel.
// d(b,k) = ||p||^2 - 2 p.c + ||c||^2 ; argmin over k needs only (-2 p.c + ||c||^2).
// GEMM done as split-bf16: p ~ ph+pl, c ~ ch+cl ; p.c ~ ph.ch + ph.cl + pl.ch.
// LDS holds separate hi/lo bf16 planes in WMMA-fragment element order, so a
// fragment is just 2 aligned ds_load_b128 with zero unpack VALU.
__global__ __launch_bounds__(NTHR) void assign_kernel(
        const float* __restrict__ points,
        const float* __restrict__ centroids,
        const float* __restrict__ c2,
        float* __restrict__ out_assign,
        float* __restrict__ out_dist,
        float* __restrict__ disp,
        float* __restrict__ newn,
        int* __restrict__ assign_i) {
    __shared__ uint32_t s_pts_hi[TILE_M * D_DIM / 2];   // bf16 pairs, 32KB
    __shared__ uint32_t s_pts_lo[TILE_M * D_DIM / 2];   // 32KB
    __shared__ uint32_t s_cent_hi[16 * D_DIM / 2];      // 8KB
    __shared__ uint32_t s_cent_lo[16 * D_DIM / 2];      // 8KB
    __shared__ float    s_x2[TILE_M];

    const int tid  = threadIdx.x;
    const int lane = tid & 31;
    const int wave = tid >> 5;          // 0..3, each wave owns 16 point rows
    const int h    = lane >> 4;         // half of wave
    const int ln   = lane & 15;
    const int rowBase = blockIdx.x * TILE_M;

    if (tid < TILE_M) s_x2[tid] = 0.f;
    __syncthreads();

    // stage the point tile: split bf16 into hi/lo planes + ||p||^2 via ds_add_f32
    for (int i = tid; i < TILE_M * D_DIM / 2; i += NTHR) {
        const int row = i >> 7;              // D/2 = 128 pairs per row
        const float2 f2 =
            *(const float2*)&points[(size_t)(rowBase + row) * D_DIM + ((i & 127) << 1)];
        uint32_t h0 = bf16_rne(f2.x), h1 = bf16_rne(f2.y);
        float hf0 = __uint_as_float(h0 << 16), hf1 = __uint_as_float(h1 << 16);
        uint32_t l0 = bf16_rne(f2.x - hf0), l1 = bf16_rne(f2.y - hf1);
        s_pts_hi[i] = h0 | (h1 << 16);
        s_pts_lo[i] = l0 | (l1 << 16);
        atomicAdd(&s_x2[row], f2.x * f2.x + f2.y * f2.y);
    }
    __syncthreads();

    float minv[8];
    int   mini[8];
#pragma unroll
    for (int j = 0; j < 8; ++j) { minv[j] = 3.4e38f; mini[j] = 0; }

    const uint16_t* arow_hi = (const uint16_t*)s_pts_hi + (wave * 16 + ln) * D_DIM;
    const uint16_t* arow_lo = (const uint16_t*)s_pts_lo + (wave * 16 + ln) * D_DIM;
    const uint16_t* brow_hi = (const uint16_t*)s_cent_hi + ln * D_DIM;
    const uint16_t* brow_lo = (const uint16_t*)s_cent_lo + ln * D_DIM;

    for (int nt = 0; nt < K_CLU / 16; ++nt) {
        __syncthreads();
        // cooperatively stage this 16-centroid tile (split bf16 planes)
        for (int i = tid; i < 16 * D_DIM / 2; i += NTHR) {
            const int row = i >> 7;
            const float2 f2 =
                *(const float2*)&centroids[(size_t)(nt * 16 + row) * D_DIM + ((i & 127) << 1)];
            uint32_t h0 = bf16_rne(f2.x), h1 = bf16_rne(f2.y);
            float hf0 = __uint_as_float(h0 << 16), hf1 = __uint_as_float(h1 << 16);
            uint32_t l0 = bf16_rne(f2.x - hf0), l1 = bf16_rne(f2.y - hf1);
            s_cent_hi[i] = h0 | (h1 << 16);
            s_cent_lo[i] = l0 | (l1 << 16);
        }
        if (nt + 1 < K_CLU / 16)
            __builtin_prefetch(&centroids[(size_t)(nt + 1) * 16 * D_DIM + tid * 8], 0, 0);
        __syncthreads();

        v8f acc = {0.f, 0.f, 0.f, 0.f, 0.f, 0.f, 0.f, 0.f};
#pragma unroll
        for (int kc = 0; kc < D_DIM; kc += 32) {
            // A fragment (16-bit A 16x32): lane (h,ln) -> row m=ln,
            // elements 0..15 contiguous at K=kc+16h.
            const v4i* pah = (const v4i*)(arow_hi + kc + 16 * h);  // 16B aligned
            const v4i* pal = (const v4i*)(arow_lo + kc + 16 * h);
            Q2 qah; qah.a = pah[0]; qah.b = pah[1];                // +16B = +8 elems
            Q2 qal; qal.a = pal[0]; qal.b = pal[1];
            // B fragment (16-bit B 32x16): lane (h,ln) -> col n=ln,
            // elements 0..15 contiguous at K=kc+16h.
            const v8i* pbh = (const v8i*)(brow_hi + kc + 16 * h);  // 32B aligned
            const v8i* pbl = (const v8i*)(brow_lo + kc + 16 * h);
            v16bf AH = __builtin_bit_cast(v16bf, qah);
            v16bf AL = __builtin_bit_cast(v16bf, qal);
            v16bf BH = __builtin_bit_cast(v16bf, pbh[0]);
            v16bf BL = __builtin_bit_cast(v16bf, pbl[0]);

            acc = __builtin_amdgcn_wmma_f32_16x16x32_bf16(false, AH, false, BH,
                                                          (short)0, acc, false, false);
            acc = __builtin_amdgcn_wmma_f32_16x16x32_bf16(false, AH, false, BL,
                                                          (short)0, acc, false, false);
            acc = __builtin_amdgcn_wmma_f32_16x16x32_bf16(false, AL, false, BH,
                                                          (short)0, acc, false, false);
        }
        // C layout: vgpr j of lane -> M = j + 8h, N = ln (same N for all 8)
        const int n = nt * 16 + ln;
        const float c2n = c2[n];
#pragma unroll
        for (int j = 0; j < 8; ++j) {
            float d = c2n - 2.0f * acc[j];
            if (d < minv[j]) { minv[j] = d; mini[j] = n; }
        }
    }

    // argmin across the 16 lanes of each half (masks 1,2,4,8 never cross halves)
#pragma unroll
    for (int j = 0; j < 8; ++j) {
        float v = minv[j];
        int   ix = mini[j];
#pragma unroll
        for (int m = 1; m < 16; m <<= 1) {
            float ov = __shfl_xor(v, m, 32);
            int   oi = __shfl_xor(ix, m, 32);
            if (ov < v || (ov == v && oi < ix)) { v = ov; ix = oi; }
        }
        if (ln == 0) {
            const int M   = j + 8 * h;
            const int row = rowBase + wave * 16 + M;
            float dist = fmaxf(v + s_x2[wave * 16 + M], 0.0f);
            out_assign[row] = (float)ix;
            out_dist[row]   = dist;
            assign_i[row]   = ix;
            atomicAdd(&disp[ix], dist);
            atomicAdd(&newn[ix], 1.0f);
        }
    }
}

// ---------------------------------------------------------------------
// new_sum[a] += points[row]  (L2-resident 1MB destination)
__global__ void scatter_points_kernel(const float* __restrict__ points,
                                      const int* __restrict__ assign_i,
                                      float* __restrict__ new_sum) {
    size_t gid = (size_t)blockIdx.x * 256 + threadIdx.x;
    int row = (int)(gid >> 8);
    int d   = (int)(gid & 255);
    int a   = assign_i[row];
    atomicAdd(&new_sum[(size_t)a * D_DIM + d], points[gid]);
}

// ---------------------------------------------------------------------
// fused: cosine distance to assigned label profile + new_labels scatter.
// labels (276 MB) streamed exactly once; running_labels/new_labels are L2-resident.
__global__ __launch_bounds__(256) void labels_kernel(
        const float* __restrict__ labels,
        const float* __restrict__ running_labels,
        const float* __restrict__ rln,
        const int* __restrict__ assign_i,
        float* __restrict__ out_dlab,
        float* __restrict__ disp_lab,
        float* __restrict__ new_labels) {
    __shared__ float rdot[256], rn2[256];
    const int row = blockIdx.x, t = threadIdx.x;
    const int a = assign_i[row];
    const float* lab = labels + (size_t)row * L_LAB;
    const float* cl  = running_labels + (size_t)a * L_LAB;
    float dot = 0.f, n2 = 0.f;
    for (int i = t; i < L_LAB; i += 256) {
        float lv = lab[i];
        dot += lv * cl[i];
        n2  += lv * lv;
        atomicAdd(&new_labels[(size_t)a * L_LAB + i], lv);
    }
    rdot[t] = dot; rn2[t] = n2;
    __syncthreads();
    for (int off = 128; off > 0; off >>= 1) {
        if (t < off) { rdot[t] += rdot[t + off]; rn2[t] += rn2[t + off]; }
        __syncthreads();
    }
    if (t == 0) {
        float dl = 1.0f - rdot[0] / (sqrtf(rn2[0]) * rln[a] + EPSV);
        out_dlab[row] = dl;
        atomicAdd(&disp_lab[a], dl);
    }
}

// ---------------------------------------------------------------------
extern "C" void kernel_launch(void* const* d_in, const int* in_sizes, int n_in,
                              void* d_out, int out_size, void* d_ws, size_t ws_size,
                              hipStream_t stream) {
    const float* points         = (const float*)d_in[0];
    const float* labels         = (const float*)d_in[1];
    const float* centroids      = (const float*)d_in[2];
    const float* running_labels = (const float*)d_in[3];

    const int Bn = B_PTS, K = K_CLU, D = D_DIM;
    float* out        = (float*)d_out;
    float* o_assign   = out;                                   // (B,)
    float* o_dist     = out + Bn;                              // (B,)
    float* o_disp     = out + 2 * (size_t)Bn;                  // (K,)
    float* o_dlab     = out + 2 * (size_t)Bn + K;              // (B,)
    float* o_displab  = out + 3 * (size_t)Bn + K;              // (K,)
    float* o_newsum   = out + 3 * (size_t)Bn + 2 * K;          // (K,D)
    float* o_newn     = o_newsum + (size_t)K * D;              // (K,)
    float* o_newlab   = o_newn + K;                            // (K,L)

    int*   assign_i = (int*)d_ws;                              // B ints
    float* c2       = (float*)((char*)d_ws + (size_t)Bn * sizeof(int));
    float* rln      = c2 + K;

    zero_kernel<<<(out_size + 255) / 256, 256, 0, stream>>>(out, out_size);
    norms_kernel<<<K, 256, 0, stream>>>(centroids, running_labels, c2, rln);
    assign_kernel<<<Bn / TILE_M, NTHR, 0, stream>>>(
        points, centroids, c2, o_assign, o_dist, o_disp, o_newn, assign_i);
    scatter_points_kernel<<<(unsigned)(((size_t)Bn * D) / 256), 256, 0, stream>>>(
        points, assign_i, o_newsum);
    labels_kernel<<<Bn, 256, 0, stream>>>(
        labels, running_labels, rln, assign_i, o_dlab, o_displab, o_newlab);
}
